// LBGC_v4_82377472737493
// MI455X (gfx1250) — matrix operand order repaired and checked by reading, not compiled, not run on hardware
//
#include <hip/hip_runtime.h>
#include <hip/hip_bf16.h>
#include <math.h>

// ---------------------------------------------------------------------------
// Problem constants (from the reference)
// ---------------------------------------------------------------------------
#define BATCH     4096
#define NSAMP     10
#define TOTAL     (BATCH * (NSAMP + 1))   // 45056 flattened samples
#define NT        168                     // unique time indices
#define DD        128                     // embedding dim (D == DT == 128)
#define MTILE     32                      // samples per block (2 WMMA row-tiles)
#define MAX_TILES 1572                    // >= ceil((TOTAL + NT*(MTILE-1))/MTILE)
#define PROJ_ELEMS (NT * DD * DD)         // 2,752,512

typedef __attribute__((ext_vector_type(16))) _Float16 v16h;
typedef __attribute__((ext_vector_type(8)))  float    v8f;

// -log_sigmoid(x) = log(1 + exp(-x)) = log1p(exp(-|x|)) - min(x, 0)
__device__ __forceinline__ float neg_log_sigmoid(float x) {
    return log1pf(expf(-fabsf(x))) - fminf(x, 0.0f);
}

__device__ __forceinline__ int sample_t(int sid, const int* __restrict__ pos_t,
                                        const int* __restrict__ neg_t) {
    return (sid < BATCH) ? pos_t[sid] : neg_t[sid - BATCH];
}

// ---------------------------------------------------------------------------
// Pass 0: pre-convert the 168 projection matrices to f16 (once per launch).
// Only 10.5 MB of unique f32 data -> 5.5 MB f16; removes all B-side
// v_cvt_pk_f16_f32 from the GEMM hot loop and halves its L2 traffic.
// ---------------------------------------------------------------------------
__global__ void cvt_proj_kernel(const float* __restrict__ src,
                                _Float16* __restrict__ dst) {
    int i = (blockIdx.x * blockDim.x + threadIdx.x) * 4;
    if (i >= PROJ_ELEMS) return;
    #pragma unroll
    for (int j = 0; j < 4; ++j) dst[i + j] = (_Float16)src[i + j];
}

// ---------------------------------------------------------------------------
// Pass 1: histogram of samples over time buckets
// ---------------------------------------------------------------------------
__global__ void count_kernel(const int* __restrict__ pos_t,
                             const int* __restrict__ neg_t,
                             int* __restrict__ count) {
    int sid = blockIdx.x * blockDim.x + threadIdx.x;
    if (sid >= TOTAL) return;
    atomicAdd(&count[sample_t(sid, pos_t, neg_t)], 1);
}

// ---------------------------------------------------------------------------
// Pass 2: MTILE-aligned exclusive scan over 168 buckets + tile->t map.
// ---------------------------------------------------------------------------
__global__ void scan_kernel(const int* __restrict__ count,
                            int* __restrict__ wr_off,
                            int* __restrict__ tile_to_t) {
    if (threadIdx.x == 0 && blockIdx.x == 0) {
        int tile = 0;
        for (int t = 0; t < NT; ++t) {
            wr_off[t] = tile * MTILE;
            int ntile = (count[t] + MTILE - 1) / MTILE;
            for (int j = 0; j < ntile; ++j) tile_to_t[tile + j] = t;
            tile += ntile;
        }
    }
}

// ---------------------------------------------------------------------------
// Pass 3: scatter sample ids into per-t, MTILE-padded buckets
// ---------------------------------------------------------------------------
__global__ void scatter_kernel(const int* __restrict__ pos_t,
                               const int* __restrict__ neg_t,
                               int* __restrict__ wr_off,
                               int* __restrict__ sorted) {
    int sid = blockIdx.x * blockDim.x + threadIdx.x;
    if (sid >= TOTAL) return;
    int t = sample_t(sid, pos_t, neg_t);
    int p = atomicAdd(&wr_off[t], 1);
    sorted[p] = sid;
}

// ---------------------------------------------------------------------------
// Pass 4: per-tile WMMA GEMM + scoring.
// One block = 32 samples sharing the same time index t.
// 8 wave32's: wave w holds ONE B fragment per k-step (P_t rows 16w..16w+15)
// and reuses it across TWO A row-tiles -> 8x v_wmma_f32_16x16x32_f16 per wave.
// PF16: B loaded from pre-converted f16 proj; else direct f32 + convert.
// ---------------------------------------------------------------------------
template <bool PF16>
__global__ __launch_bounds__(256) void
tile_gemm_kernel(const int* __restrict__ sorted,
                 const int* __restrict__ tile_to_t,
                 const int* __restrict__ pos_u, const int* __restrict__ pos_p,
                 const int* __restrict__ neg_u, const int* __restrict__ neg_p,
                 const float* __restrict__ user_emb,
                 const float* __restrict__ poi_emb,
                 const float* __restrict__ time_emb,
                 const float* __restrict__ proj_f32,
                 const _Float16* __restrict__ proj_f16,
                 float* __restrict__ out) {
    __shared__ _Float16 s_uh[MTILE][DD];   // U tile in f16 (A operand source)
    __shared__ float    s_up[MTILE][DD];   // UP = U @ P^T result
    __shared__ float    s_te[DD];          // time embedding row
    __shared__ int   s_uid[MTILE], s_pid[MTILE], s_ns[MTILE],
                     s_dst[MTILE], s_valid[MTILE];
    __shared__ float s_acc[MTILE];

    const int tile = blockIdx.x;
    const int t = tile_to_t[tile];
    if (t < 0) return;                  // unused tile slot (uniform exit)

    const int tid  = threadIdx.x;
    const int lane = tid & 31;
    const int wave = tid >> 5;

    // ---- Phase 0: decode the 32 sample ids of this tile -------------------
    if (tid < MTILE) {
        int sid = sorted[tile * MTILE + tid];
        int uid = 0, pid = 0, ns = -1, dst = 0, valid = 0;
        if (sid >= 0) {
            valid = 1;
            if (sid < BATCH) {              // positive branch
                uid = pos_u[sid]; pid = pos_p[sid]; dst = sid; ns = -1;
            } else {                        // negative branch: sid = B + s*B + b
                int k = sid - BATCH;
                uid = neg_u[k]; pid = neg_p[k]; ns = k >> 12;  // k / 4096
            }
        }
        s_uid[tid] = uid; s_pid[tid] = pid; s_ns[tid] = ns;
        s_dst[tid] = dst; s_valid[tid] = valid; s_acc[tid] = 0.0f;
    }
    __syncthreads();

    // ---- Phase 1: stage gathered user rows (f32 -> f16) + time row --------
    {
        int r  = tid >> 3;            // 0..31 : sample row
        int c0 = (tid & 7) * 16;      // 16 elems per thread
        const float* urow = user_emb + (size_t)s_uid[r] * DD;
        #pragma unroll
        for (int j = 0; j < 16; ++j) s_uh[r][c0 + j] = (_Float16)urow[c0 + j];
        if (tid < DD) s_te[tid] = time_emb[(size_t)t * DD + tid];
    }
    __syncthreads();

    // ---- Phase 2: WMMA. EXEC all-1s, no divergence. -----------------------
    {
        const int m    = lane & 15;
        const int selA = (lane < 16) ? 0 : 8;    // A: interleaved K pattern
        const int selB = (lane < 16) ? 0 : 8;    // B: interleaved K pattern (matches A)
        // B^T row n of this wave's stripe == P_t row (16*wave + n)
        const size_t prow_ofs = (size_t)t * (DD * DD)
                              + (size_t)(wave * 16 + (lane & 15)) * DD;
        const float*    prow_f = proj_f32 + prow_ofs;
        const _Float16* prow_h = proj_f16 + prow_ofs;
        v8f acc0 = {}, acc1 = {};
        #pragma unroll
        for (int kk = 0; kk < 4; ++kk) {
            const int kb = kk * 32;
            v16h a0, a1, b;
            #pragma unroll
            for (int h = 0; h < 16; ++h)
                b[h] = PF16 ? prow_h[kb + selB + h + (h & 8)]
                            : (_Float16)prow_f[kb + selB + h + (h & 8)];
            #pragma unroll
            for (int h = 0; h < 8; ++h) {
                a0[h]     = s_uh[m][kb + selA + h];
                a0[8 + h] = s_uh[m][kb + selA + 16 + h];
                a1[h]     = s_uh[16 + m][kb + selA + h];
                a1[8 + h] = s_uh[16 + m][kb + selA + 16 + h];
            }
            acc0 = __builtin_amdgcn_wmma_f32_16x16x32_f16(
                false, a0, false, b, (short)0, acc0, false, false);
            acc1 = __builtin_amdgcn_wmma_f32_16x16x32_f16(
                false, a1, false, b, (short)0, acc1, false, false);
        }
        // C/D layout: VGPR r -> M=r (lanes 0-15) / M=8+r (lanes 16-31), N=lane%16
        const int n    = lane & 15;
        const int mofs = (lane < 16) ? 0 : 8;
        #pragma unroll
        for (int r = 0; r < 8; ++r) {
            s_up[mofs + r][wave * 16 + n]      = acc0[r];
            s_up[16 + mofs + r][wave * 16 + n] = acc1[r];
        }
    }
    __syncthreads();

    // ---- Phase 3: score_b = sum_i (UP[b,i] + te[i]) * poi[p_b, i] ---------
    {
        int s  = tid & 31;    // sample
        int ib = tid >> 5;    // 8 threads/sample, 16 strided elems each
        if (s_valid[s]) {
            const float* prow_p = poi_emb + (size_t)s_pid[s] * DD;
            float partial = 0.0f;
            #pragma unroll
            for (int j = 0; j < 16; ++j) {
                int i = ib + 8 * j;
                partial += (s_up[s][i] + s_te[i]) * prow_p[i];
            }
            atomicAdd(&s_acc[s], partial);   // ds_add_f32
        }
    }
    __syncthreads();

    // ---- Phase 4: loss + output writes ------------------------------------
    if (tid < MTILE && s_valid[tid]) {
        float sc = s_acc[tid];
        int ns = s_ns[tid];
        if (ns < 0) out[s_dst[tid]] = neg_log_sigmoid(sc);            // pos[b]
        else atomicAdd(&out[BATCH + ns], neg_log_sigmoid(-sc));       // neg[s]
    }
}

// ---------------------------------------------------------------------------
// Host-side launch (graph-capture safe: only async ops on `stream`)
// ---------------------------------------------------------------------------
extern "C" void kernel_launch(void* const* d_in, const int* in_sizes, int n_in,
                              void* d_out, int out_size, void* d_ws, size_t ws_size,
                              hipStream_t stream) {
    const int*   pos_u    = (const int*)d_in[0];
    const int*   pos_t    = (const int*)d_in[1];
    const int*   pos_p    = (const int*)d_in[2];
    const int*   neg_u    = (const int*)d_in[3];
    const int*   neg_t    = (const int*)d_in[4];
    const int*   neg_p    = (const int*)d_in[5];
    // d_in[6] = NS scalar (hardcoded NSAMP=10)
    const float* user_emb = (const float*)d_in[7];
    const float* poi_emb  = (const float*)d_in[8];
    const float* time_emb = (const float*)d_in[9];
    const float* proj_emb = (const float*)d_in[10];
    float* out = (float*)d_out;

    // Workspace layout: int region, then (optional) f16 projection table.
    int* ws        = (int*)d_ws;
    int* count     = ws;                            // NT
    int* wr_off    = ws + NT;                       // NT
    int* tile_to_t = ws + 2 * NT;                   // MAX_TILES
    int* sorted    = ws + 2 * NT + MAX_TILES;       // MAX_TILES * MTILE
    size_t int_bytes = (size_t)(2 * NT + MAX_TILES + MAX_TILES * MTILE)
                     * sizeof(int);
    size_t proj_off  = (int_bytes + 255) & ~(size_t)255;   // 256B align
    _Float16* proj_h = (_Float16*)((char*)d_ws + proj_off);
    const bool use_f16 = (ws_size >= proj_off + (size_t)PROJ_ELEMS * sizeof(_Float16));

    hipMemsetAsync(count, 0, NT * sizeof(int), stream);
    hipMemsetAsync(tile_to_t, 0xFF, MAX_TILES * sizeof(int), stream);        // -1
    hipMemsetAsync(sorted, 0xFF, MAX_TILES * MTILE * sizeof(int), stream);   // -1
    hipMemsetAsync(out, 0, (BATCH + NSAMP) * sizeof(float), stream);

    const int nblk = (TOTAL + 255) / 256;
    count_kernel  <<<nblk, 256, 0, stream>>>(pos_t, neg_t, count);
    scan_kernel   <<<1, 32, 0, stream>>>(count, wr_off, tile_to_t);
    scatter_kernel<<<nblk, 256, 0, stream>>>(pos_t, neg_t, wr_off, sorted);

    if (use_f16) {
        cvt_proj_kernel<<<(PROJ_ELEMS / 4 + 255) / 256, 256, 0, stream>>>(
            proj_emb, proj_h);
        tile_gemm_kernel<true><<<MAX_TILES, 256, 0, stream>>>(
            sorted, tile_to_t, pos_u, pos_p, neg_u, neg_p,
            user_emb, poi_emb, time_emb, proj_emb, proj_h, out);
    } else {
        tile_gemm_kernel<false><<<MAX_TILES, 256, 0, stream>>>(
            sorted, tile_to_t, pos_u, pos_p, neg_u, neg_p,
            user_emb, poi_emb, time_emb, proj_emb, (const _Float16*)nullptr, out);
    }
}
